// GraphCastEncoder_58007828299996
// MI455X (gfx1250) — matrix operand order repaired and checked by reading, not compiled, not run on hardware
//
#include <hip/hip_runtime.h>

typedef __bf16 bf16;
typedef __attribute__((ext_vector_type(16))) bf16  v16bf;
typedef __attribute__((ext_vector_type(8)))  bf16  v8bf;
typedef __attribute__((ext_vector_type(8)))  float v8f;
typedef __attribute__((ext_vector_type(4)))  float v4f;

#define HDIM   512
#define NGRID  65536
#define NMESH  10242
#define NEDGE  131072

// ---------------------------------------------------------------------------
// CDNA5 async global->LDS copy (16 bytes per lane), tracked by ASYNCcnt.
// LDS operand = workgroup-relative LDS byte offset = low 32 bits of the flat
// shared-aperture address (ISA 10.2).
// ---------------------------------------------------------------------------
__device__ __forceinline__ void async_copy16(const void* lds_dst, const void* gsrc) {
    unsigned lo = (unsigned)(unsigned long long)lds_dst;
    asm volatile("global_load_async_to_lds_b128 %0, %1, off"
                 :: "v"(lo), "v"(gsrc) : "memory");
}
__device__ __forceinline__ void wait_async0() {
    asm volatile("s_wait_asynccnt 0x0" ::: "memory");
}

// ---------------------------------------------------------------------------
// Weight convert + transpose via LDS tiles (coalesced both sides):
//   W [K][HDIM] fp32 row-major  ->  Wt [HDIM][K] bf16
// ---------------------------------------------------------------------------
__global__ __launch_bounds__(256)
void wconv_kernel(const float* __restrict__ W, bf16* __restrict__ Wt, int K) {
    __shared__ float t[64][65];
    const int ktiles = K >> 6;
    const int bk = blockIdx.x % ktiles;
    const int bn = blockIdx.x / ktiles;
    const int k0 = bk * 64, n0 = bn * 64;
    const int r = threadIdx.x >> 2;          // 0..63
    const int c = (threadIdx.x & 3) * 16;    // 0,16,32,48
    #pragma unroll
    for (int j = 0; j < 16; j += 4) {
        v4f f = *reinterpret_cast<const v4f*>(W + (size_t)(k0 + r) * HDIM + n0 + c + j);
        t[r][c + j + 0] = f[0]; t[r][c + j + 1] = f[1];
        t[r][c + j + 2] = f[2]; t[r][c + j + 3] = f[3];
    }
    __syncthreads();
    v16bf o;
    #pragma unroll
    for (int j = 0; j < 16; ++j) o[j] = (bf16)t[c + j][r];
    *reinterpret_cast<v16bf*>(Wt + (size_t)(n0 + r) * K + k0 + c) = o;
}

// ---------------------------------------------------------------------------
// GEMM1: h = SiLU(A @ W1 + b1), h stored bf16.
// A rows: concat of up to 3 segments of width 512 (edge | grid[src] | mesh[dst]).
// Block tile 128x128, 8 waves (2x4), wave tile 64x32. Double-buffered LDS.
// B staged via async global->LDS; A staged via load+cvt (fp32 -> bf16) through
// a single running pointer (segment switch = uniform cold branch, no selects).
// ---------------------------------------------------------------------------
__global__ __launch_bounds__(256)
void gemm1_silu(const float* __restrict__ s0,
                const float* __restrict__ s1,
                const float* __restrict__ s2,
                const int*   __restrict__ i1,
                const int*   __restrict__ i2,
                const bf16*  __restrict__ Wt,     // [HDIM][Ktot] (N-major)
                const float* __restrict__ bias,   // [HDIM]
                bf16*        __restrict__ hout,   // [rows][HDIM]
                int rows, int Ktot)
{
    __shared__ __align__(64) bf16 As[2][128 * 32];
    __shared__ __align__(64) bf16 Bs[2][128 * 32];   // Bs[n][k]

    const int tid  = threadIdx.x;
    const int lane = tid & 31;
    const int wid  = tid >> 5;
    const int wr   = wid >> 2;        // 0..1
    const int wc   = wid & 3;         // 0..3
    const int half = lane >> 4;       // 0/1
    const int l15  = lane & 15;

    const int nblocks = HDIM / 128;   // 4
    const int bm   = blockIdx.x / nblocks;
    const int bn   = blockIdx.x % nblocks;
    const int row0 = bm * 128;
    const int col0 = bn * 128;

    // ---- A staging coords: thread -> (row, 16-elem half), hoisted ----
    const int sr   = tid >> 1;                 // 0..127
    const int skk  = (tid & 1) * 16;           // 0 or 16
    const int sgr  = row0 + sr;
    const int sgrc = (sgr < rows) ? sgr : (rows - 1);   // clamp; stores are guarded
    const float* a0 = s0 + (size_t)sgrc * HDIM + skk;
    const float* a1 = s1 ? (s1 + (size_t)(i1 ? i1[sgrc] : sgrc) * HDIM + skk) : a0;
    const float* a2 = s2 ? (s2 + (size_t)(i2 ? i2[sgrc] : sgrc) * HDIM + skk) : a0;

    // ---- B staging coords: 16B chunks; chunk c -> lds byte offset c*16 ----
    const int bn8 = tid >> 2;                  // 0..63
    const int bk8 = (tid & 3) * 8;
    const bf16* brow0 = Wt + (size_t)(col0 + bn8) * Ktot + bk8;
    const bf16* brow1 = brow0 + (size_t)64 * Ktot;

    auto stageA_at = [&](const float* p, int buf) {
        v16bf av;
        #pragma unroll
        for (int v = 0; v < 4; ++v) {
            v4f f = *reinterpret_cast<const v4f*>(p + v * 4);
            av[v * 4 + 0] = (bf16)f[0]; av[v * 4 + 1] = (bf16)f[1];
            av[v * 4 + 2] = (bf16)f[2]; av[v * 4 + 3] = (bf16)f[3];
        }
        *reinterpret_cast<v16bf*>(&As[buf][sr * 32 + skk]) = av;
    };
    auto stageB = [&](int kt, int buf) {
        const int k0 = kt << 5;
        async_copy16(&Bs[buf][(size_t)tid * 8],         brow0 + k0);
        async_copy16(&Bs[buf][(size_t)(tid + 256) * 8], brow1 + k0);
    };

    v8f acc[4][2];
    #pragma unroll
    for (int mt = 0; mt < 4; ++mt)
        #pragma unroll
        for (int nt = 0; nt < 2; ++nt)
            #pragma unroll
            for (int e = 0; e < 8; ++e) acc[mt][nt][e] = 0.0f;

    const int kiters = Ktot >> 5;
    stageA_at(a0, 0);
    stageB(0, 0);
    wait_async0();
    __syncthreads();

    // ap = A source for the NEXT tile to stage (tile kt+1 at top of iter kt)
    const float* ap = a0 + 32;

    for (int kt = 0; kt < kiters; ++kt) {
        const int cur = kt & 1;
        const int nxt = cur ^ 1;
        if (kt + 1 < kiters) {
            stageB(kt + 1, nxt);
            stageA_at(ap, nxt);
            __builtin_prefetch(ap + 32, 0, 3);
        }

        v16bf aF[4], bF[2];
        #pragma unroll
        for (int mt = 0; mt < 4; ++mt) {
            const bf16* ab = &As[cur][(wr * 64 + mt * 16 + l15) * 32];
            aF[mt] = *reinterpret_cast<const v16bf*>(ab + half * 16);
        }
        #pragma unroll
        for (int nt = 0; nt < 2; ++nt)
            bF[nt] = *reinterpret_cast<const v16bf*>(
                &Bs[cur][(wc * 32 + nt * 16 + l15) * 32 + half * 16]);
        #pragma unroll
        for (int mt = 0; mt < 4; ++mt)
            #pragma unroll
            for (int nt = 0; nt < 2; ++nt)
                acc[mt][nt] = __builtin_amdgcn_wmma_f32_16x16x32_bf16(
                    false, aF[mt], false, bF[nt], (short)0, acc[mt][nt], false, false);

        wait_async0();
        __syncthreads();

        // advance A pointer to tile kt+2; uniform cold branch at segment entry
        ap += 32;
        if (((kt + 2) & 15) == 0) {
            ap = (((kt + 2) >> 4) == 1) ? a1 : a2;
        }
    }

    // ---- epilogue: SiLU(acc + bias) -> bf16 ----
    #pragma unroll
    for (int mt = 0; mt < 4; ++mt) {
        #pragma unroll
        for (int nt = 0; nt < 2; ++nt) {
            const int col = col0 + wc * 32 + nt * 16 + l15;
            const float bv = bias[col];
            #pragma unroll
            for (int j = 0; j < 8; ++j) {
                const int gr = row0 + wr * 64 + mt * 16 + half * 8 + j;
                if (gr < rows) {
                    float x = acc[mt][nt][j] + bv;
                    float y = x / (1.0f + __expf(-x));
                    hout[(size_t)gr * HDIM + col] = (bf16)y;
                }
            }
        }
    }
}

// ---------------------------------------------------------------------------
// GEMM2: x = h @ W2 + b2 ; y = resid + LN(x)*gamma + beta
//   grid/mesh mode: write y to outp
//   edge mode     : atomic scatter-add y into agg[dst[row]]  (segment_sum)
// Block tile 32x512 (whole rows -> in-kernel LayerNorm), 8 waves side by side,
// wave tile 32x64. Double-buffered async staging; 64KB fp32 LN tile aliased.
// ---------------------------------------------------------------------------
__global__ __launch_bounds__(256)
void gemm2_ln(const bf16*  __restrict__ hin,    // [rows][HDIM] bf16
              const bf16*  __restrict__ Wt,     // [HDIM][HDIM] (N-major)
              const float* __restrict__ b2,
              const float* __restrict__ gamma,
              const float* __restrict__ beta,
              const float* __restrict__ resid,  // [rows][HDIM]
              float*       __restrict__ outp,   // null in edge mode
              const int*   __restrict__ dst,    // non-null => edge mode
              float*       __restrict__ agg,    // [NMESH][HDIM], edge mode
              int rows)
{
    // staging: As[2] (2KB each) + Bs[2] (32KB each) = 68KB; LN tile 64KB aliased
    __shared__ __align__(64) char smem[69632];
    bf16* As0 = reinterpret_cast<bf16*>(smem);
    bf16* As1 = reinterpret_cast<bf16*>(smem + 2048);
    bf16* Bs0 = reinterpret_cast<bf16*>(smem + 4096);
    bf16* Bs1 = reinterpret_cast<bf16*>(smem + 4096 + 32768);
    float* tile = reinterpret_cast<float*>(smem);          // 32*512 f32 (aliased)

    const int tid  = threadIdx.x;
    const int lane = tid & 31;
    const int wid  = tid >> 5;        // 0..7 -> column strip wid*64
    const int half = lane >> 4;
    const int l15  = lane & 15;
    const int row0 = blockIdx.x * 32;

    // ---- A staging: 128 chunks of 16B; thread t<128 -> row t>>2, k8 (t&3)*8 ----
    const int aw   = tid >> 2;                 // 0..63 (rows valid for tid<128)
    const int ak8  = (tid & 3) * 8;
    const int sgr  = row0 + (aw & 31);
    const int sgrc = (sgr < rows) ? sgr : (rows - 1);
    const bf16* arow = hin + (size_t)sgrc * HDIM + ak8;
    // ---- B staging: 2048 chunks of 16B; 8 per thread ----
    const bf16* wrow = Wt + (size_t)(tid >> 2) * HDIM + ak8;

    auto stageA = [&](int kt, bf16* Asb) {
        if (tid < 128)
            async_copy16(&Asb[(size_t)(aw & 31) * 32 + ak8], arow + (kt << 5));
    };
    auto stageB = [&](int kt, bf16* Bsb) {
        const int k0 = kt << 5;
        #pragma unroll
        for (int i = 0; i < 8; ++i)
            async_copy16(&Bsb[(size_t)(tid + i * 256) * 8],
                         wrow + (size_t)i * 64 * HDIM + k0);
    };

    v8f acc[2][4];
    #pragma unroll
    for (int mt = 0; mt < 2; ++mt)
        #pragma unroll
        for (int nt = 0; nt < 4; ++nt)
            #pragma unroll
            for (int e = 0; e < 8; ++e) acc[mt][nt][e] = 0.0f;

    const int kiters = HDIM >> 5;     // 16
    stageA(0, As0);
    stageB(0, Bs0);
    wait_async0();
    __syncthreads();

    for (int kt = 0; kt < kiters; ++kt) {
        bf16* Asc = (kt & 1) ? As1 : As0;
        bf16* Bsc = (kt & 1) ? Bs1 : Bs0;
        if (kt + 1 < kiters) {
            bf16* Asn = (kt & 1) ? As0 : As1;
            bf16* Bsn = (kt & 1) ? Bs0 : Bs1;
            stageB(kt + 1, Bsn);
            stageA(kt + 1, Asn);
        }
        if (kt + 2 < kiters) __builtin_prefetch(arow + ((kt + 2) << 5), 0, 3);

        v16bf aF[2], bF[4];
        #pragma unroll
        for (int mt = 0; mt < 2; ++mt) {
            const bf16* ab = &Asc[(mt * 16 + l15) * 32];
            aF[mt] = *reinterpret_cast<const v16bf*>(ab + half * 16);
        }
        #pragma unroll
        for (int nt = 0; nt < 4; ++nt)
            bF[nt] = *reinterpret_cast<const v16bf*>(
                &Bsc[(wid * 64 + nt * 16 + l15) * 32 + half * 16]);
        #pragma unroll
        for (int mt = 0; mt < 2; ++mt)
            #pragma unroll
            for (int nt = 0; nt < 4; ++nt)
                acc[mt][nt] = __builtin_amdgcn_wmma_f32_16x16x32_bf16(
                    false, aF[mt], false, bF[nt], (short)0, acc[mt][nt], false, false);

        wait_async0();
        __syncthreads();
    }

    // ---- park acc + bias into fp32 LDS tile (aliases staging) ----
    #pragma unroll
    for (int mt = 0; mt < 2; ++mt) {
        #pragma unroll
        for (int nt = 0; nt < 4; ++nt) {
            const int col = wid * 64 + nt * 16 + l15;
            const float bv = b2[col];
            #pragma unroll
            for (int j = 0; j < 8; ++j) {
                const int r = mt * 16 + half * 8 + j;
                tile[r * HDIM + col] = acc[mt][nt][j] + bv;
            }
        }
    }
    __syncthreads();

    // ---- LayerNorm + residual (+ scatter-sum in edge mode) ----
    for (int rr = 0; rr < 4; ++rr) {
        const int r  = wid * 4 + rr;
        const int gr = row0 + r;
        float s = 0.0f, q = 0.0f;
        #pragma unroll
        for (int i = 0; i < 16; ++i) {
            float x = tile[r * HDIM + lane + i * 32];
            s += x; q += x * x;
        }
        #pragma unroll
        for (int off = 16; off >= 1; off >>= 1) {
            s += __shfl_xor(s, off);
            q += __shfl_xor(q, off);
        }
        const float mean = s * (1.0f / 512.0f);
        const float var  = q * (1.0f / 512.0f) - mean * mean;
        const float rstd = rsqrtf(var + 1e-5f);
        if (gr < rows) {
            if (dst) {
                const int d = dst[gr];
                #pragma unroll
                for (int i = 0; i < 16; ++i) {
                    const int col = lane + i * 32;
                    const float y = resid[(size_t)gr * HDIM + col] +
                                    (tile[r * HDIM + col] - mean) * rstd * gamma[col] + beta[col];
                    atomicAdd(&agg[(size_t)d * HDIM + col], y);
                }
            } else {
                #pragma unroll
                for (int i = 0; i < 16; ++i) {
                    const int col = lane + i * 32;
                    outp[(size_t)gr * HDIM + col] =
                        resid[(size_t)gr * HDIM + col] +
                        (tile[r * HDIM + col] - mean) * rstd * gamma[col] + beta[col];
                }
            }
        }
    }
}

// ---------------------------------------------------------------------------
extern "C" void kernel_launch(void* const* d_in, const int* in_sizes, int n_in,
                              void* d_out, int out_size, void* d_ws, size_t ws_size,
                              hipStream_t stream)
{
    (void)in_sizes; (void)n_in; (void)out_size; (void)ws_size;

    const float* grid_f = (const float*)d_in[0];
    const float* mesh_f = (const float*)d_in[1];
    const float* edge_f = (const float*)d_in[2];
    const int*   src    = (const int*)d_in[3];
    const int*   dstI   = (const int*)d_in[4];
    const float* We1 = (const float*)d_in[5];
    const float* be1 = (const float*)d_in[6];
    const float* We2 = (const float*)d_in[7];
    const float* be2 = (const float*)d_in[8];
    const float* ge  = (const float*)d_in[9];
    const float* bbe = (const float*)d_in[10];
    const float* Wn1 = (const float*)d_in[11];
    const float* bn1 = (const float*)d_in[12];
    const float* Wn2 = (const float*)d_in[13];
    const float* bn2 = (const float*)d_in[14];
    const float* gn  = (const float*)d_in[15];
    const float* bbn = (const float*)d_in[16];
    const float* Wg1 = (const float*)d_in[17];
    const float* bg1 = (const float*)d_in[18];
    const float* Wg2 = (const float*)d_in[19];
    const float* bg2 = (const float*)d_in[20];
    const float* gg  = (const float*)d_in[21];
    const float* bbg = (const float*)d_in[22];

    // ---- workspace layout ----
    const size_t HH = (size_t)HDIM * HDIM;
    bf16* Wt_e1 = (bf16*)d_ws;                 // [H][3H]
    bf16* Wt_e2 = Wt_e1 + 3 * HH;              // [H][H]
    bf16* Wt_n1 = Wt_e2 + HH;                  // [H][2H]
    bf16* Wt_n2 = Wt_n1 + 2 * HH;              // [H][H]
    bf16* Wt_g1 = Wt_n2 + HH;                  // [H][H]
    bf16* Wt_g2 = Wt_g1 + HH;                  // [H][H]
    float* agg  = (float*)(Wt_g2 + HH);        // [NMESH][H] fp32
    bf16* hbuf  = (bf16*)(agg + (size_t)NMESH * HDIM);   // [max(E,NGRID)][H] bf16

    // ---- weight convert/transpose (bf16, [N][K]) ----
    wconv_kernel<<<dim3((3 * HDIM / 64) * (HDIM / 64)), 256, 0, stream>>>(We1, Wt_e1, 3 * HDIM);
    wconv_kernel<<<dim3((    HDIM / 64) * (HDIM / 64)), 256, 0, stream>>>(We2, Wt_e2, HDIM);
    wconv_kernel<<<dim3((2 * HDIM / 64) * (HDIM / 64)), 256, 0, stream>>>(Wn1, Wt_n1, 2 * HDIM);
    wconv_kernel<<<dim3((    HDIM / 64) * (HDIM / 64)), 256, 0, stream>>>(Wn2, Wt_n2, HDIM);
    wconv_kernel<<<dim3((    HDIM / 64) * (HDIM / 64)), 256, 0, stream>>>(Wg1, Wt_g1, HDIM);
    wconv_kernel<<<dim3((    HDIM / 64) * (HDIM / 64)), 256, 0, stream>>>(Wg2, Wt_g2, HDIM);
    hipMemsetAsync(agg, 0, (size_t)NMESH * HDIM * sizeof(float), stream);

    float* out_grid = (float*)d_out;
    float* out_mesh = out_grid + (size_t)NGRID * HDIM;

    // ---- MeshEdgeBlock: edge MLP + residual, scatter-sum into agg ----
    gemm1_silu<<<dim3((NEDGE / 128) * (HDIM / 128)), 256, 0, stream>>>(
        edge_f, grid_f, mesh_f, src, dstI, Wt_e1, be1, hbuf, NEDGE, 3 * HDIM);
    gemm2_ln<<<dim3(NEDGE / 32), 256, 0, stream>>>(
        hbuf, Wt_e2, be2, ge, bbe, edge_f, nullptr, dstI, agg, NEDGE);

    // ---- MeshNodeBlock: cat(mesh, agg) MLP + residual ----
    gemm1_silu<<<dim3(((NMESH + 127) / 128) * (HDIM / 128)), 256, 0, stream>>>(
        mesh_f, agg, nullptr, nullptr, nullptr, Wt_n1, bn1, hbuf, NMESH, 2 * HDIM);
    gemm2_ln<<<dim3((NMESH + 31) / 32), 256, 0, stream>>>(
        hbuf, Wt_n2, bn2, gn, bbn, mesh_f, out_mesh, nullptr, nullptr, NMESH);

    // ---- Grid node MLP + residual ----
    gemm1_silu<<<dim3((NGRID / 128) * (HDIM / 128)), 256, 0, stream>>>(
        grid_f, nullptr, nullptr, nullptr, nullptr, Wt_g1, bg1, hbuf, NGRID, HDIM);
    gemm2_ln<<<dim3(NGRID / 32), 256, 0, stream>>>(
        hbuf, Wt_g2, bg2, gg, bbg, grid_f, out_grid, nullptr, nullptr, NGRID);
}
